// ScoreBlock_5222680232109
// MI455X (gfx1250) — compile-verified, matrix operand order checked by reading, not run on hardware
//
#include <hip/hip_runtime.h>
#include <hip/hip_bf16.h>

// Sizes from the reference
#define Bn 8
#define PB 8192
#define PS 8192
#define Cc 256
#define NP 32
#define Mh 16          // NP/2 pointers actually used
#define KTOP 1024      // PS/8

typedef __attribute__((ext_vector_type(16))) _Float16 v16h;
typedef __attribute__((ext_vector_type(8)))  float    v8f;

// ---------------------------------------------------------------------------
// Zero-fill the huge `selected` region (256 MB). Pure bandwidth.
// ---------------------------------------------------------------------------
__global__ __launch_bounds__(256) void k_zero(float4* __restrict__ p, long long n4) {
    long long i = (long long)blockIdx.x * blockDim.x + threadIdx.x;
    long long stride = (long long)gridDim.x * blockDim.x;
    for (; i < n4; i += stride) p[i] = make_float4(0.f, 0.f, 0.f, 0.f);
}

// ---------------------------------------------------------------------------
// kernels[b,c] = masked mean of gathered x_b rows; also k_norm, has_any -> ws
// One block per batch, 256 threads (one per channel).
// ---------------------------------------------------------------------------
__global__ __launch_bounds__(256) void k_kernels(const float* __restrict__ xb,
                                                 const int*   __restrict__ bidx,
                                                 float* __restrict__ out_kern,
                                                 float* __restrict__ wsf) {
    int b = blockIdx.x, c = threadIdx.x;
    float sum = 0.f; int cnt = 0;
#pragma unroll
    for (int j = 0; j < Mh; ++j) {
        int id = bidx[b * NP + j];
        if (id >= 0) { sum += xb[((size_t)b * PB + id) * Cc + c]; cnt++; }
    }
    float kv = sum / (float)(cnt > 0 ? cnt : 1);
    out_kern[b * Cc + c] = kv;

    __shared__ float red[256];
    red[c] = kv * kv;
    __syncthreads();
#pragma unroll
    for (int s = 128; s > 0; s >>= 1) {
        if (c < s) red[c] += red[c + s];
        __syncthreads();
    }
    if (c == 0) {
        wsf[b]      = fmaxf(sqrtf(red[0]), 1e-8f);   // clamped k_norm
        wsf[Bn + b] = (cnt > 0) ? 1.f : 0.f;         // has_any
    }
}

// ---------------------------------------------------------------------------
// pos_scores via WMMA. 8 waves/block, each wave owns a 16-row tile of x_s.
// A = 16x32 f16 tile of x_s (8 K-steps over C=256), B = kernel vector
// broadcast into all 16 columns. Row norms accumulated in-flight.
// ---------------------------------------------------------------------------
__global__ __launch_bounds__(256) void k_scores(const float* __restrict__ xs,
                                                const float* __restrict__ kern,
                                                const float* __restrict__ wsf,
                                                float* __restrict__ pos_scores) {
    int b     = blockIdx.x >> 6;        // 64 tile-blocks per batch
    int tblk  = blockIdx.x & 63;
    int wave  = threadIdx.x >> 5;
    int lane  = threadIdx.x & 31;
    int hi16  = lane >> 4;
    int row   = (tblk * 8 + wave) * 16 + (lane & 15);

    __shared__ _Float16 kh[Cc];         // kernel vector in f16
    __shared__ float dots[8][16];       // per-wave D column-0 exchange
    kh[threadIdx.x] = (_Float16)kern[b * Cc + threadIdx.x];
    __syncthreads();

    const float* xrow = xs + ((size_t)b * PS + row) * Cc;

    v8f acc = {};
    float sumsq = 0.f;
#pragma unroll
    for (int kc = 0; kc < 8; ++kc) {
        // ISA A-frag layout (16-bit 16x32): lane holds K = {K0..K0+7, K1..K1+7}
        int K0 = kc * 32 + hi16 * 8;
        int K1 = K0 + 16;
        float4 p0 = *(const float4*)(xrow + K0);
        float4 p1 = *(const float4*)(xrow + K0 + 4);
        float4 p2 = *(const float4*)(xrow + K1);
        float4 p3 = *(const float4*)(xrow + K1 + 4);

        sumsq += p0.x*p0.x + p0.y*p0.y + p0.z*p0.z + p0.w*p0.w
               + p1.x*p1.x + p1.y*p1.y + p1.z*p1.z + p1.w*p1.w
               + p2.x*p2.x + p2.y*p2.y + p2.z*p2.z + p2.w*p2.w
               + p3.x*p3.x + p3.y*p3.y + p3.z*p3.z + p3.w*p3.w;

        v16h a, bf;
        a[0]=(_Float16)p0.x; a[1]=(_Float16)p0.y; a[2]=(_Float16)p0.z; a[3]=(_Float16)p0.w;
        a[4]=(_Float16)p1.x; a[5]=(_Float16)p1.y; a[6]=(_Float16)p1.z; a[7]=(_Float16)p1.w;
        a[8]=(_Float16)p2.x; a[9]=(_Float16)p2.y; a[10]=(_Float16)p2.z; a[11]=(_Float16)p2.w;
        a[12]=(_Float16)p3.x; a[13]=(_Float16)p3.y; a[14]=(_Float16)p3.z; a[15]=(_Float16)p3.w;
#pragma unroll
        for (int h = 0; h < 8; ++h) { bf[h] = kh[K0 + h]; bf[8 + h] = kh[K1 + h]; }

        // D = A x B + C : all 16 columns identical (kernel broadcast)
        acc = __builtin_amdgcn_wmma_f32_16x16x32_f16(
                  false, a, false, bf, (short)0, acc, false, false);
    }

    // full row norm: lane L holds half the channels, L^16 the other half
    sumsq += __shfl_xor(sumsq, 16, 32);

    // D column 0: rows 0-7 in lane 0 (VGPR 0-7), rows 8-15 in lane 16
    if (lane == 0) {
#pragma unroll
        for (int r = 0; r < 8; ++r) dots[wave][r] = acc[r];
    }
    if (lane == 16) {
#pragma unroll
        for (int r = 0; r < 8; ++r) dots[wave][8 + r] = acc[r];
    }
    __syncthreads();

    if (lane < 16) {
        float dot     = dots[wave][lane];
        float xs_norm = fmaxf(sqrtf(sumsq), 1e-8f);
        float knorm   = wsf[b];
        float hasany  = wsf[Bn + b];
        float cosv    = dot / (xs_norm * knorm);
        pos_scores[(size_t)b * PS + row] = (cosv + 1.f) * 0.5f * hasany;
    }
}

// ---------------------------------------------------------------------------
// Top-k (1024 of 8192) per batch via descending bitonic sort of packed keys
// in 64 KB LDS (CDNA5 WGP has 320 KB). key = score_bits<<32 | (8191-idx):
// scores are >= 0 so float bits are order-preserving; low bits reproduce
// jax.lax.top_k's ascending-index tie-break. Then scatter 1.0 into the
// pre-zeroed `selected` region (straight-through binary output).
// ---------------------------------------------------------------------------
__global__ __launch_bounds__(1024) void k_topk(const float* __restrict__ pos_scores,
                                               float* __restrict__ out_sel,
                                               float* __restrict__ out_idx) {
    int b = blockIdx.x, tid = threadIdx.x;
    __shared__ unsigned long long keys[PS];   // 64 KB

    for (int i = tid; i < PS; i += 1024) {
        float s = pos_scores[(size_t)b * PS + i];
        unsigned int sb = __float_as_uint(s);
        keys[i] = ((unsigned long long)sb << 32) | (unsigned int)(PS - 1 - i);
    }
    __syncthreads();

    for (int k = 2; k <= PS; k <<= 1) {
        for (int j = k >> 1; j > 0; j >>= 1) {
            for (int i = tid; i < PS; i += 1024) {
                int ixj = i ^ j;
                if (ixj > i) {
                    unsigned long long a = keys[i], c = keys[ixj];
                    bool up = ((i & k) == 0);
                    if (up ? (a < c) : (a > c)) { keys[i] = c; keys[ixj] = a; }
                }
            }
            __syncthreads();
        }
    }

    unsigned long long key = keys[tid];           // tid-th largest
    unsigned int sb = (unsigned int)(key >> 32);
    int idx = (PS - 1) - (int)(key & 0xFFFFFFFFull);
    out_idx[b * KTOP + tid] = (float)idx;
    if (sb != 0u)                                  // topk_score > 0 -> binary 1.0
        out_sel[((size_t)b * KTOP + tid) * PS + idx] = 1.0f;
}

// ---------------------------------------------------------------------------
extern "C" void kernel_launch(void* const* d_in, const int* in_sizes, int n_in,
                              void* d_out, int out_size, void* d_ws, size_t ws_size,
                              hipStream_t stream) {
    const float* x_b  = (const float*)d_in[0];
    const float* x_s  = (const float*)d_in[1];
    const int*   bidx = (const int*)  d_in[2];
    float* out = (float*)d_out;

    // flat output layout (floats), in reference return order
    const size_t SEL = (size_t)Bn * KTOP * PS;   // 67,108,864
    const size_t IDX = (size_t)Bn * KTOP;        //      8,192
    const size_t PSC = (size_t)Bn * PS;          //     65,536
    const size_t XSZ = (size_t)Bn * PS * Cc;     // 16,777,216
    float* o_sel = out;
    float* o_idx = o_sel + SEL;
    float* o_ps  = o_idx + IDX;
    float* o_xs  = o_ps  + PSC;
    float* o_krn = o_xs  + XSZ;
    float* wsf   = (float*)d_ws;                 // [0..7]=k_norm, [8..15]=has_any

    // pass-through x_s (device-to-device, capture-safe)
    hipMemcpyAsync(o_xs, x_s, XSZ * sizeof(float), hipMemcpyDeviceToDevice, stream);

    k_zero   <<<8192, 256, 0, stream>>>((float4*)o_sel, (long long)(SEL / 4));
    k_kernels<<<Bn,   256, 0, stream>>>(x_b, bidx, o_krn, wsf);
    k_scores <<<Bn * 64, 256, 0, stream>>>(x_s, o_krn, wsf, o_ps);
    k_topk   <<<Bn,  1024, 0, stream>>>(o_ps, o_sel, o_idx);
}
